// WordAttn_76811195122199
// MI455X (gfx1250) — compile-verified, matrix operand order checked by reading, not run on hardware
//
#include <hip/hip_runtime.h>
#include <hip/hip_bf16.h>

// ---------------------------------------------------------------------------
// Bidirectional GRU + word attention for MI455X (gfx1250), f16 WMMA / f32 acc.
// T=512, B=512, E=128, H=64, V=50001
// ---------------------------------------------------------------------------

typedef _Float16 h8   __attribute__((ext_vector_type(8)));
typedef _Float16 v16h __attribute__((ext_vector_type(16)));
typedef float    v8f  __attribute__((ext_vector_type(8)));

#define T_LEN 512
#define B_SZ  512
#define E_DIM 128
#define H_DIM 64

static __device__ __forceinline__ v8f splat8(float v) {
    v8f r = {v, v, v, v, v, v, v, v};
    return r;
}
static __device__ __forceinline__ float fsigmoid(float x) {
    return 1.0f / (1.0f + __expf(-x));
}
static __device__ __forceinline__ float ftanh(float x) {
    return 2.0f / (1.0f + __expf(-2.0f * x)) - 1.0f;
}

// Load one 16x32 f16 fragment (A layout, or B layout from B^T rows).
// `tile` points at (row 0, k 0); `ld` = row stride in halves.
// Per ISA: lane L -> row L%16, K chunks {kb, kb+16}, kb = (L/16)*8.
static __device__ __forceinline__ v16h load_frag(const _Float16* tile, int ld) {
    const int lane = threadIdx.x & 31;
    const int r    = lane & 15;
    const int kb   = (lane >> 4) << 3;
    const _Float16* p = tile + r * ld + kb;
    union { v16h v; h8 h[2]; } f;
    f.h[0] = *(const h8*)(p);
    f.h[1] = *(const h8*)(p + 16);
    return f.v;
}

static __device__ __forceinline__ v8f wmma_f16(v16h a, v16h b, v8f c) {
    return __builtin_amdgcn_wmma_f32_16x16x32_f16(false, a, false, b,
                                                  (short)0, c, false, false);
}

// --------------------------- prep: weight converts --------------------------

// w_ih_f [192,128] ++ w_ih_b [192,128] -> f16 [384,128]
__global__ void cvt_wih(const float* wf, const float* wb, _Float16* o) {
    int i = blockIdx.x * 256 + threadIdx.x;
    if (i >= 384 * 128) return;
    o[i] = (i < 192 * 128) ? (_Float16)wf[i] : (_Float16)wb[i - 192 * 128];
}

// w_hh_f [192,64] ++ w_hh_b [192,64] -> f16 [2][192][64]
__global__ void cvt_whh(const float* wf, const float* wb, _Float16* o) {
    int i = blockIdx.x * 256 + threadIdx.x;
    if (i >= 2 * 192 * 64) return;
    o[i] = (i < 192 * 64) ? (_Float16)wf[i] : (_Float16)wb[i - 192 * 64];
}

// word_weight [k=128][n=128] -> transposed f16 [n][k] (B^T row-major)
__global__ void cvt_ww(const float* w, _Float16* o) {
    int i = blockIdx.x * 256 + threadIdx.x;
    if (i >= 128 * 128) return;
    int n = i >> 7, k = i & 127;
    o[i] = (_Float16)w[k * 128 + n];
}

// --------------------------- embedding gather -------------------------------

__global__ __launch_bounds__(128) void embed_gather(const int* tok,
                                                    const float* embed,
                                                    _Float16* x16) {
    int row = blockIdx.x;           // 0 .. T*B-1  (t*B + b)
    int c   = threadIdx.x;          // 0 .. 127
    int tk  = tok[row];
    x16[(size_t)row * E_DIM + c] = (_Float16)embed[(size_t)tk * E_DIM + c];
}

// --------------------------- phase 1: xg = x @ Wih^T + b --------------------
// M = T*B = 262144, N = 384 (fwd r,z,n | bwd r,z,n), K = 128.
// One wave per block; each block does one 16x64 output tile.
// B fragments for a whole N-tile are preloaded so the 8 b128 loads issue as
// one clause and overlap the WMMA accumulation chain.

__global__ __launch_bounds__(32) void xg_gemm(const _Float16* x16,
                                              const _Float16* wih16,
                                              const float* bih_f,
                                              const float* bih_b,
                                              float* xg) {
    const int mt   = blockIdx.x;          // 0..16383
    const int nb   = blockIdx.y;          // 0..5 -> cols nb*64 .. +63
    const int lane = threadIdx.x & 31;
    const int nlo  = lane & 15;
    const int mb   = (lane >> 4) << 3;    // row base for D fragment

    v16h A[4];
    const _Float16* ab = x16 + (size_t)mt * 16 * E_DIM;
    #pragma unroll
    for (int kk = 0; kk < 4; ++kk) A[kk] = load_frag(ab + kk * 32, E_DIM);

    #pragma unroll
    for (int jt = 0; jt < 4; ++jt) {
        const int n0 = nb * 64 + jt * 16;
        const int ng = n0 + nlo;                       // global col 0..383
        const float bv = (ng < 192) ? bih_f[ng] : bih_b[ng - 192];

        // preload all 4 k-step B fragments for this N-tile
        v16h Bt[4];
        #pragma unroll
        for (int kk = 0; kk < 4; ++kk)
            Bt[kk] = load_frag(wih16 + (size_t)n0 * E_DIM + kk * 32, E_DIM);

        v8f D = splat8(bv);
        #pragma unroll
        for (int kk = 0; kk < 4; ++kk) D = wmma_f16(A[kk], Bt[kk], D);

        float* op = xg + ((size_t)mt * 16 + mb) * 384 + ng;
        #pragma unroll
        for (int r = 0; r < 8; ++r) op[(size_t)r * 384] = D[r];
    }
}

// --------------------------- phase 2: recurrent scan ------------------------
// grid = (B/16, 2 dirs), block = 128 (4 waves). Wave j owns h-cols j*16..+15
// and computes gate N-tiles {j, j+4, j+8} (r, z, n) so gate math stays local.
// xg rows for step t+1 are prefetched (global_prefetch) during step t to hide
// HBM latency in this inherently serial loop.

__global__ __launch_bounds__(128) void gru_scan(const float* hidden0,
                                                const float* xg,
                                                const float* bhh_f,
                                                const float* bhh_b,
                                                const _Float16* whh16,
                                                _Float16* ht,
                                                float* out_hn) {
    __shared__ __align__(16) _Float16 hbuf[2][16][72];  // double-buffered h tile

    const int dir   = blockIdx.y;
    const int btile = blockIdx.x;           // 0..31
    const int wave  = threadIdx.x >> 5;     // j = 0..3
    const int lane  = threadIdx.x & 31;
    const int nlo   = lane & 15;
    const int mb    = (lane >> 4) << 3;     // D row base AND A k-chunk base (==kb)
    const int col   = wave * 16 + nlo;      // h column 0..63

    const float* bhh = dir ? bhh_b : bhh_f;

    // Loop-invariant B fragments: w_hh^T rows, gates r/z/n, ksteps 0/1.
    v16h Bf[3][2];
    const _Float16* wbase = whh16 + (size_t)dir * 192 * H_DIM;
    #pragma unroll
    for (int g = 0; g < 3; ++g)
        #pragma unroll
        for (int kk = 0; kk < 2; ++kk)
            Bf[g][kk] = load_frag(wbase + (size_t)(g * 64 + wave * 16) * H_DIM + kk * 32, H_DIM);

    v8f Cb[3];
    #pragma unroll
    for (int g = 0; g < 3; ++g) Cb[g] = splat8(bhh[g * 64 + col]);

    // init h from hidden_state [2,B,H]
    float hreg[8];
    #pragma unroll
    for (int r = 0; r < 8; ++r) {
        hreg[r] = hidden0[(size_t)dir * B_SZ * H_DIM +
                          (size_t)(btile * 16 + mb + r) * H_DIM + col];
        hbuf[0][mb + r][col] = (_Float16)hreg[r];
    }

    // prefetch first step's xg rows
    {
        const int tx0 = dir ? (T_LEN - 1) : 0;
        const float* xp0 = xg + ((size_t)tx0 * B_SZ + btile * 16 + mb) * 384 +
                           dir * 192 + col;
        #pragma unroll
        for (int r = 0; r < 8; ++r) {
            const float* rowp = xp0 + (size_t)r * 384;
            __builtin_prefetch(rowp, 0, 1);
            __builtin_prefetch(rowp + 64, 0, 1);
            __builtin_prefetch(rowp + 128, 0, 1);
        }
    }
    __syncthreads();

    for (int t = 0; t < T_LEN; ++t) {
        const int tx  = dir ? (T_LEN - 1 - t) : t;
        const int cur = t & 1, nxt = cur ^ 1;

        // A fragments from LDS h (M=16, K=64 -> 2 ksteps)
        union { v16h v; h8 h[2]; } fa;
        fa.h[0] = *(const h8*)&hbuf[cur][nlo][mb];
        fa.h[1] = *(const h8*)&hbuf[cur][nlo][mb + 16];
        v16h A0 = fa.v;
        fa.h[0] = *(const h8*)&hbuf[cur][nlo][mb + 32];
        fa.h[1] = *(const h8*)&hbuf[cur][nlo][mb + 48];
        v16h A1 = fa.v;

        v8f D[3];
        #pragma unroll
        for (int g = 0; g < 3; ++g) {
            D[g] = wmma_f16(A0, Bf[g][0], Cb[g]);
            D[g] = wmma_f16(A1, Bf[g][1], D[g]);
        }

        // prefetch next step's xg rows while WMMAs are in flight
        if (t + 1 < T_LEN) {
            const int txn = dir ? (T_LEN - 2 - t) : (t + 1);
            const float* xpn = xg + ((size_t)txn * B_SZ + btile * 16 + mb) * 384 +
                               dir * 192 + col;
            #pragma unroll
            for (int r = 0; r < 8; ++r) {
                const float* rowp = xpn + (size_t)r * 384;
                __builtin_prefetch(rowp, 0, 1);
                __builtin_prefetch(rowp + 64, 0, 1);
                __builtin_prefetch(rowp + 128, 0, 1);
            }
        }

        // xg fragment loads + GRU gate math
        const float* xp = xg + ((size_t)tx * B_SZ + btile * 16 + mb) * 384 +
                          dir * 192 + col;
        _Float16* htp = ht + ((size_t)tx * B_SZ + btile * 16 + mb) * 128 +
                        dir * H_DIM + col;
        #pragma unroll
        for (int r = 0; r < 8; ++r) {
            const float* rowp = xp + (size_t)r * 384;
            float xr = rowp[0], xz = rowp[64], xn = rowp[128];
            float rv = fsigmoid(xr + D[0][r]);
            float zv = fsigmoid(xz + D[1][r]);
            float nv = ftanh(xn + rv * D[2][r]);
            float hn = (1.0f - zv) * nv + zv * hreg[r];
            hreg[r] = hn;
            hbuf[nxt][mb + r][col] = (_Float16)hn;
            htp[(size_t)r * 128] = (_Float16)hn;
        }
        __syncthreads();
    }

    // h_n [2,B,H]
    #pragma unroll
    for (int r = 0; r < 8; ++r)
        out_hn[(size_t)dir * B_SZ * H_DIM +
               (size_t)(btile * 16 + mb + r) * H_DIM + col] = hreg[r];
}

// --------------------------- phase 3a: attention scores ---------------------
// scores[m] = (tanh(h_t[m,:] @ W + b)) . ctx ; W via WMMA, dot fused in epilogue.

__global__ __launch_bounds__(32) void attn_scores(const _Float16* ht,
                                                  const _Float16* ww16,
                                                  const float* wbias,
                                                  const float* ctx,
                                                  float* scores) {
    const int mt   = blockIdx.x;          // 0..16383
    const int lane = threadIdx.x & 31;
    const int nlo  = lane & 15;
    const int mb   = (lane >> 4) << 3;

    v16h A[4];
    const _Float16* ab = ht + (size_t)mt * 16 * 128;
    #pragma unroll
    for (int kk = 0; kk < 4; ++kk) A[kk] = load_frag(ab + kk * 32, 128);

    float sacc[8];
    #pragma unroll
    for (int r = 0; r < 8; ++r) sacc[r] = 0.0f;

    for (int nt = 0; nt < 8; ++nt) {
        const int n = nt * 16 + nlo;

        v16h Bt[4];
        #pragma unroll
        for (int kk = 0; kk < 4; ++kk)
            Bt[kk] = load_frag(ww16 + (size_t)(nt * 16) * 128 + kk * 32, 128);

        v8f D = splat8(wbias[n]);
        #pragma unroll
        for (int kk = 0; kk < 4; ++kk) D = wmma_f16(A[kk], Bt[kk], D);

        const float cv = ctx[n];
        #pragma unroll
        for (int r = 0; r < 8; ++r) sacc[r] += ftanh(D[r]) * cv;
    }

    // reduce the 16 N-lanes within each half-wave
    #pragma unroll
    for (int off = 1; off < 16; off <<= 1)
        #pragma unroll
        for (int r = 0; r < 8; ++r)
            sacc[r] += __shfl_xor(sacc[r], off, 32);

    if (nlo == 0) {
        #pragma unroll
        for (int r = 0; r < 8; ++r) scores[mt * 16 + mb + r] = sacc[r];
    }
}

// --------------------------- phase 3b: softmax over T + weighted sum --------

__global__ __launch_bounds__(128) void attn_out(const float* scores,
                                                const _Float16* ht,
                                                float* out) {
    const int b   = blockIdx.x;      // 0..511
    const int tid = threadIdx.x;     // 0..127
    __shared__ float p[T_LEN];
    __shared__ float red[128];

    float lm = -3.4e38f;
    for (int t = tid; t < T_LEN; t += 128) {
        float s = scores[(size_t)t * B_SZ + b];
        p[t] = s;
        lm = fmaxf(lm, s);
    }
    red[tid] = lm;
    __syncthreads();
    for (int s = 64; s > 0; s >>= 1) {
        if (tid < s) red[tid] = fmaxf(red[tid], red[tid + s]);
        __syncthreads();
    }
    const float mx = red[0];
    float ls = 0.0f;
    for (int t = tid; t < T_LEN; t += 128) {
        float e = __expf(p[t] - mx);
        p[t] = e;
        ls += e;
    }
    __syncthreads();
    red[tid] = ls;
    __syncthreads();
    for (int s = 64; s > 0; s >>= 1) {
        if (tid < s) red[tid] += red[tid + s];
        __syncthreads();
    }
    const float sum = red[0];
    __syncthreads();

    // thread tid = channel c; s_i[b,c] = sum_t p[t]*h_t[t,b,c] / sum
    float acc = 0.0f;
    const _Float16* hp = ht + (size_t)b * 128 + tid;
    for (int t = 0; t < T_LEN; ++t)
        acc += p[t] * (float)hp[(size_t)t * B_SZ * 128];
    out[(size_t)b * 128 + tid] = acc / sum;
}

// --------------------------- launcher ---------------------------------------

extern "C" void kernel_launch(void* const* d_in, const int* in_sizes, int n_in,
                              void* d_out, int out_size, void* d_ws, size_t ws_size,
                              hipStream_t stream) {
    const int*   tok    = (const int*)d_in[0];
    const float* hid0   = (const float*)d_in[1];
    const float* embed  = (const float*)d_in[2];
    const float* wword  = (const float*)d_in[3];
    const float* wbias  = (const float*)d_in[4];
    const float* ctx    = (const float*)d_in[5];
    const float* w_ih_f = (const float*)d_in[6];
    const float* w_hh_f = (const float*)d_in[7];
    const float* b_ih_f = (const float*)d_in[8];
    const float* b_hh_f = (const float*)d_in[9];
    const float* w_ih_b = (const float*)d_in[10];
    const float* w_hh_b = (const float*)d_in[11];
    const float* b_ih_b = (const float*)d_in[12];
    const float* b_hh_b = (const float*)d_in[13];

    float* out = (float*)d_out;            // [s_i: 65536][h_n: 65536]

    // workspace layout (bytes)
    char* ws = (char*)d_ws;
    _Float16* x16    = (_Float16*)(ws + 0);                       //  64 MB
    float*    xg     = (float*)   (ws + 67108864ull);             // 384 MB
    _Float16* ht     = (_Float16*)(ws + 469762048ull);            //  64 MB
    float*    scores = (float*)   (ws + 536870912ull);            //   1 MB
    _Float16* wih16  = (_Float16*)(ws + 537919488ull);            //  96 KB
    _Float16* whh16  = (_Float16*)(ws + 538017792ull);            //  48 KB
    _Float16* ww16   = (_Float16*)(ws + 538066944ull);            //  32 KB

    // weight conversion
    cvt_wih<<<(384 * 128 + 255) / 256, 256, 0, stream>>>(w_ih_f, w_ih_b, wih16);
    cvt_whh<<<(2 * 192 * 64 + 255) / 256, 256, 0, stream>>>(w_hh_f, w_hh_b, whh16);
    cvt_ww <<<(128 * 128 + 255) / 256, 256, 0, stream>>>(wword, ww16);

    // embedding gather -> f16
    embed_gather<<<T_LEN * B_SZ, 128, 0, stream>>>(tok, embed, x16);

    // xg = x @ Wih^T + b_ih  (both directions, N=384)
    xg_gemm<<<dim3(T_LEN * B_SZ / 16, 6), 32, 0, stream>>>(x16, wih16,
                                                           b_ih_f, b_ih_b, xg);

    // recurrent scan (writes h_t and h_n)
    gru_scan<<<dim3(B_SZ / 16, 2), 128, 0, stream>>>(hid0, xg, b_hh_f, b_hh_b,
                                                     whh16, ht, out + 65536);

    // attention scores, then softmax + weighted sum -> s_i
    attn_scores<<<T_LEN * B_SZ / 16, 32, 0, stream>>>(ht, ww16, wbias, ctx, scores);
    attn_out<<<B_SZ, 128, 0, stream>>>(scores, ht, out);
}
